// EPLBRouter_24215025615339
// MI455X (gfx1250) — compile-verified, matrix operand order checked by reading, not run on hardware
//
#include <hip/hip_runtime.h>
#include <hip/hip_bf16.h>

typedef __attribute__((ext_vector_type(16))) _Float16 v16h;
typedef __attribute__((ext_vector_type(8)))  float    v8f;
typedef __attribute__((ext_vector_type(4)))  unsigned int v4u;
typedef int v4i __attribute__((__vector_size__(16)));

#define EPLB_GLOBAL_AS __attribute__((address_space(1)))
#define EPLB_LDS_AS    __attribute__((address_space(3)))

union FragV { v16h h; v4u u[2]; };
union Chunk { v4u u; _Float16 h[8]; };

#if __has_builtin(__builtin_amdgcn_global_load_async_to_lds_b128)
#define EPLB_ASYNC_LDS 1
#else
#define EPLB_ASYNC_LDS 0
#endif

__device__ __forceinline__ void eplb_wait_async0() {
#if EPLB_ASYNC_LDS
#if __has_builtin(__builtin_amdgcn_s_wait_asynccnt)
    __builtin_amdgcn_s_wait_asynccnt(0);
#else
    asm volatile("s_wait_asynccnt 0" ::: "memory");
#endif
#endif
}

// ---------------------------------------------------------------------------
// fp32 -> f16 elementwise convert (weights)
// ---------------------------------------------------------------------------
__global__ void eplb_cvt_f16(const float* __restrict__ in, _Float16* __restrict__ out, int n) {
    int i = blockIdx.x * 256 + threadIdx.x;
    if (i < n) out[i] = (_Float16)in[i];
}

// ---------------------------------------------------------------------------
// LayerNorm per token (H=2048), block = 256 threads (8 waves), f16 output.
// Each thread owns 8 contiguous elements: b128 loads, one b128 store.
// ---------------------------------------------------------------------------
__global__ void eplb_ln_f16(const float* __restrict__ x,
                            const float* __restrict__ gamma,
                            const float* __restrict__ beta,
                            _Float16* __restrict__ out, int H) {
    const int tok = blockIdx.x;
    const float* xr = x + (size_t)tok * H;
    const int i0 = threadIdx.x * 8;
    float4 a = *(const float4*)(xr + i0);
    float4 b = *(const float4*)(xr + i0 + 4);
    float v[8] = {a.x, a.y, a.z, a.w, b.x, b.y, b.z, b.w};
    float s = 0.f, s2 = 0.f;
#pragma unroll
    for (int i = 0; i < 8; ++i) { s += v[i]; s2 += v[i] * v[i]; }
    for (int o = 16; o > 0; o >>= 1) {
        s  += __shfl_xor(s,  o, 32);
        s2 += __shfl_xor(s2, o, 32);
    }
    __shared__ float sh[16];
    const int wave = threadIdx.x >> 5, lane = threadIdx.x & 31;
    if (lane == 0) { sh[wave] = s; sh[8 + wave] = s2; }
    __syncthreads();
    float S = 0.f, S2 = 0.f;
#pragma unroll
    for (int w = 0; w < 8; ++w) { S += sh[w]; S2 += sh[8 + w]; }
    const float mean = S / (float)H;
    const float var  = S2 / (float)H - mean * mean;
    const float rstd = rsqrtf(var + 1e-5f);
    float4 g0 = *(const float4*)(gamma + i0);
    float4 g1 = *(const float4*)(gamma + i0 + 4);
    float4 b0 = *(const float4*)(beta + i0);
    float4 b1 = *(const float4*)(beta + i0 + 4);
    float g[8] = {g0.x, g0.y, g0.z, g0.w, g1.x, g1.y, g1.z, g1.w};
    float be[8] = {b0.x, b0.y, b0.z, b0.w, b1.x, b1.y, b1.z, b1.w};
    Chunk o8;
#pragma unroll
    for (int i = 0; i < 8; ++i)
        o8.h[i] = (_Float16)((v[i] - mean) * rstd * g[i] + be[i]);
    *(v4u*)(out + (size_t)tok * H + i0) = o8.u;
}

// ---------------------------------------------------------------------------
// WMMA GEMM: C[M,N] = epilogue( A[M,K] @ B[K,N] + bias[N] )
//   block: 256 threads = 8 waves (2 x 4), tile 128 x BN, BK = 64 (2 wmma k-steps)
//   double-buffered LDS; A tile staged via async global->LDS when available,
//   B tile staged via registers and stored transposed.
//   __launch_bounds__(256, 1): full VGPR budget, no spills.
// ---------------------------------------------------------------------------
template <int BN>
__global__ __launch_bounds__(256, 1)
void eplb_wmma_gemm(const _Float16* __restrict__ A,
                    const _Float16* __restrict__ B,
                    const float* __restrict__ bias,
                    _Float16* __restrict__ Ch,
                    float* __restrict__ Cf,
                    int M, int N, int K,
                    float alpha, int accumulate, int relu, int out_half) {
    constexpr int BM = 128, BK = 64;
    constexpr int TN  = BN / 64;                  // wmma tiles per wave in N
    constexpr int ACH = (BM * BK) / (8 * 256);    // A 8-half chunks / thread (=4)
    constexpr int BCH = (BK * BN) / (8 * 256);    // B chunks / thread (2 or 4)
    constexpr int BNC = BN / 8;                   // B chunks per k-row
    constexpr int LDW = BK + 8;                   // 72 halfs = 144B stride (16B aligned)

    __shared__ __align__(16) _Float16 lA[2][BM][LDW];
    __shared__ __align__(16) _Float16 lBt[2][BN][LDW];   // B transposed [n][k]

    const int tid   = threadIdx.x;
    const int lane  = tid & 31;
    const int wave  = tid >> 5;
    const int wm    = wave >> 2;       // 0..1  (64 rows each)
    const int wn    = wave & 3;        // 0..3  (BN/4 cols each)
    const int lhalf = lane >> 4;       // 0/1
    const int lrow  = lane & 15;
    const int mBase = blockIdx.y * BM;
    const int nBase = blockIdx.x * BN;

    // ---- per-thread staging coordinates, 32-bit offsets from uniform bases
    int aRow[ACH], aCol[ACH];
    int aOff[ACH];                      // element offsets into A (SADDR+voffset form)
    int bKr[BCH], bCol[BCH];
    int bOff[BCH];
#pragma unroll
    for (int c = 0; c < ACH; ++c) {
        int id = tid + c * 256;
        aRow[c] = id >> 3;              // BK/8 = 8 chunks per row
        aCol[c] = (id & 7) << 3;
        aOff[c] = (mBase + aRow[c]) * K + aCol[c];
    }
#pragma unroll
    for (int c = 0; c < BCH; ++c) {
        int id = tid + c * 256;
        bKr[c]  = id / BNC;
        bCol[c] = (id % BNC) << 3;
        bOff[c] = bKr[c] * N + nBase + bCol[c];
    }

    v8f acc[4][TN];
#pragma unroll
    for (int tm = 0; tm < 4; ++tm)
#pragma unroll
        for (int tn = 0; tn < TN; ++tn)
            acc[tm][tn] = {};

#if !EPLB_ASYNC_LDS
    Chunk aReg[ACH];
#endif
    Chunk bReg[BCH];

    auto issueA = [&](int buf) {
#pragma unroll
        for (int c = 0; c < ACH; ++c) {
#if EPLB_ASYNC_LDS
            v4i* gp = (v4i*)(A + aOff[c]);                     // generic reinterpret
            v4i* lp = (v4i*)&lA[buf][aRow[c]][aCol[c]];        // generic reinterpret
            __builtin_amdgcn_global_load_async_to_lds_b128(
                (EPLB_GLOBAL_AS v4i*)gp, (EPLB_LDS_AS v4i*)lp, 0, 0);
#else
            aReg[c].u = *(const v4u*)(A + aOff[c]);
#endif
        }
    };
    auto loadB = [&]() {
#pragma unroll
        for (int c = 0; c < BCH; ++c) bReg[c].u = *(const v4u*)(B + bOff[c]);
    };
    auto advance = [&]() {
#pragma unroll
        for (int c = 0; c < ACH; ++c) aOff[c] += BK;
#pragma unroll
        for (int c = 0; c < BCH; ++c) bOff[c] += BK * N;
    };
    auto drain = [&](int buf) {
#if !EPLB_ASYNC_LDS
#pragma unroll
        for (int c = 0; c < ACH; ++c)
            *(v4u*)&lA[buf][aRow[c]][aCol[c]] = aReg[c].u;
#endif
#pragma unroll
        for (int c = 0; c < BCH; ++c)
#pragma unroll
            for (int i = 0; i < 8; ++i)
                lBt[buf][bCol[c] + i][bKr[c]] = bReg[c].h[i];
        eplb_wait_async0();
    };

    // ---- prologue: stage tile 0 into buffer 0
    issueA(0);
    loadB();
    advance();
    drain(0);
    __syncthreads();

    const int NK = K / BK;
    for (int ki = 0; ki < NK; ++ki) {
        const int buf = ki & 1;
        const bool more = (ki + 1 < NK);
        if (more) { issueA(buf ^ 1); loadB(); advance(); }

        // ---- 2 wmma k-substeps from LDS buffer `buf`
#pragma unroll
        for (int ks = 0; ks < 2; ++ks) {
            FragV fa[4];
#pragma unroll
            for (int tm = 0; tm < 4; ++tm) {
                int r = wm * 64 + tm * 16 + lrow;
                fa[tm].u[0] = *(const v4u*)&lA[buf][r][ks * 32 + 8 * lhalf];
                fa[tm].u[1] = *(const v4u*)&lA[buf][r][ks * 32 + 16 + 8 * lhalf];
            }
#pragma unroll
            for (int tn = 0; tn < TN; ++tn) {
                int cb = wn * (BN / 4) + tn * 16 + lrow;
                FragV fb;
                fb.u[0] = *(const v4u*)&lBt[buf][cb][ks * 32 + 8 * lhalf];
                fb.u[1] = *(const v4u*)&lBt[buf][cb][ks * 32 + 16 + 8 * lhalf];
#pragma unroll
                for (int tm = 0; tm < 4; ++tm)
                    acc[tm][tn] = __builtin_amdgcn_wmma_f32_16x16x32_f16(
                        false, fa[tm].h, false, fb.h, (short)0, acc[tm][tn], false, false);
            }
        }
        if (more) drain(buf ^ 1);
        __syncthreads();
    }

    // ---- epilogue: C layout = lanes 0-15 N=lane M=j, lanes 16-31 N=lane-16 M=j+8
#pragma unroll
    for (int tm = 0; tm < 4; ++tm) {
#pragma unroll
        for (int tn = 0; tn < TN; ++tn) {
            int nG = nBase + wn * (BN / 4) + tn * 16 + lrow;
            float bv = bias[nG];
#pragma unroll
            for (int j = 0; j < 8; ++j) {
                int mG = mBase + wm * 64 + tm * 16 + 8 * lhalf + j;
                float v = alpha * (acc[tm][tn][j] + bv);
                if (relu) v = fmaxf(v, 0.f);
                size_t idx = (size_t)mG * N + nG;
                if (out_half) {
                    Ch[idx] = (_Float16)v;
                } else {
                    if (accumulate) v += Cf[idx];
                    Cf[idx] = v;
                }
            }
        }
    }
}

// ---------------------------------------------------------------------------
// softmax over E=64 + top-2 + expert-load accumulation; one wave per token
// ---------------------------------------------------------------------------
__global__ void eplb_softmax_topk(const float* __restrict__ logits,
                                  int* __restrict__ outIdx,
                                  float* __restrict__ outP,
                                  float* __restrict__ loadAcc) {
    const int wave = threadIdx.x >> 5, lane = threadIdx.x & 31;
    const int tok = blockIdx.x * 8 + wave;
    const float* lg = logits + (size_t)tok * 64;
    float v0 = lg[lane * 2], v1 = lg[lane * 2 + 1];
    float mx = fmaxf(v0, v1);
    for (int o = 16; o > 0; o >>= 1) mx = fmaxf(mx, __shfl_xor(mx, o, 32));
    float e0 = __expf(v0 - mx), e1 = __expf(v1 - mx);
    float s = e0 + e1;
    for (int o = 16; o > 0; o >>= 1) s += __shfl_xor(s, o, 32);
    float inv = 1.f / s;
    float p0 = e0 * inv, p1 = e1 * inv;
    atomicAdd(&loadAcc[lane * 2], p0);
    atomicAdd(&loadAcc[lane * 2 + 1], p1);
    // local sorted pair
    float a1, a2; int i1, i2;
    if (p0 >= p1) { a1 = p0; i1 = lane * 2;     a2 = p1; i2 = lane * 2 + 1; }
    else          { a1 = p1; i1 = lane * 2 + 1; a2 = p0; i2 = lane * 2;     }
    // merge sorted pairs across the wave
    for (int o = 16; o > 0; o >>= 1) {
        float b1 = __shfl_xor(a1, o, 32), b2 = __shfl_xor(a2, o, 32);
        int  bi1 = __shfl_xor(i1, o, 32), bi2 = __shfl_xor(i2, o, 32);
        float n1, n2; int ni1, ni2;
        if (a1 >= b1) {
            n1 = a1; ni1 = i1;
            if (b1 >= a2) { n2 = b1; ni2 = bi1; } else { n2 = a2; ni2 = i2; }
        } else {
            n1 = b1; ni1 = bi1;
            if (a1 >= b2) { n2 = a1; ni2 = i1; } else { n2 = b2; ni2 = bi2; }
        }
        a1 = n1; i1 = ni1; a2 = n2; i2 = ni2;
    }
    if (lane == 0) {
        float d = 1.f / (a1 + a2);
        outIdx[tok * 2]     = i1;
        outIdx[tok * 2 + 1] = i2;
        outP[tok * 2]       = a1 * d;
        outP[tok * 2 + 1]   = a2 * d;
    }
}

__global__ void eplb_zero64(float* p) { p[threadIdx.x] = 0.f; }

__global__ void eplb_finalize(const float* __restrict__ expert_loads,
                              const float* __restrict__ loadAcc,
                              float* __restrict__ auxOut,   // [0]=aux, [1..64]=new_loads
                              float invT) {
    const int e = threadIdx.x;   // 64 threads
    __shared__ float sh[64];
    float nl = 0.9f * expert_loads[e] + 0.1f * (loadAcc[e] * invT);
    sh[e] = nl;
    auxOut[1 + e] = nl;
    __syncthreads();
    if (e == 0) {
        float tot = 0.f;
        for (int i = 0; i < 64; ++i) tot += sh[i];
        float target = tot / 64.f;
        float aux = 0.f;
        for (int i = 0; i < 64; ++i) {
            float d = sh[i] - target;
            aux += d * d;
        }
        auxOut[0] = aux;
    }
}

// ---------------------------------------------------------------------------
extern "C" void kernel_launch(void* const* d_in, const int* in_sizes, int n_in,
                              void* d_out, int out_size, void* d_ws, size_t ws_size,
                              hipStream_t stream) {
    (void)in_sizes; (void)n_in; (void)out_size; (void)ws_size;
    const float* x    = (const float*)d_in[0];
    const float* lng  = (const float*)d_in[1];
    const float* lnb  = (const float*)d_in[2];
    const float* w1   = (const float*)d_in[3];
    const float* b1   = (const float*)d_in[4];
    const float* w2   = (const float*)d_in[5];
    const float* b2   = (const float*)d_in[6];
    const float* lw1  = (const float*)d_in[7];
    const float* lb1  = (const float*)d_in[8];
    const float* lw2  = (const float*)d_in[9];
    const float* lb2  = (const float*)d_in[10];
    const float* eld  = (const float*)d_in[11];

    const int T = 4 * 4096, H = 2048, Hh = 1024, E = 64;

    char* ws = (char*)d_ws;
    size_t off = 0;
    auto alloc = [&](size_t bytes) -> void* {
        void* p = ws + off;
        off += (bytes + 255) & ~(size_t)255;
        return p;
    };
    _Float16* xn    = (_Float16*)alloc((size_t)T * H * 2);
    _Float16* w1f   = (_Float16*)alloc((size_t)H * H * 2);
    _Float16* w2f   = (_Float16*)alloc((size_t)H * E * 2);
    _Float16* lw1f  = (_Float16*)alloc((size_t)H * Hh * 2);
    _Float16* lw2f  = (_Float16*)alloc((size_t)Hh * E * 2);
    _Float16* hbuf  = (_Float16*)alloc((size_t)T * H * 2);
    _Float16* lhbuf = (_Float16*)alloc((size_t)T * Hh * 2);
    float* logits   = (float*)alloc((size_t)T * E * 4);
    float* loadAcc  = (float*)alloc((size_t)E * 4);

    // weight conversions fp32 -> f16
    eplb_cvt_f16<<<(H * H + 255) / 256, 256, 0, stream>>>(w1, w1f, H * H);
    eplb_cvt_f16<<<(H * E + 255) / 256, 256, 0, stream>>>(w2, w2f, H * E);
    eplb_cvt_f16<<<(H * Hh + 255) / 256, 256, 0, stream>>>(lw1, lw1f, H * Hh);
    eplb_cvt_f16<<<(Hh * E + 255) / 256, 256, 0, stream>>>(lw2, lw2f, Hh * E);

    // layernorm -> f16 activations
    eplb_ln_f16<<<T, 256, 0, stream>>>(x, lng, lnb, xn, H);

    // h  = relu(xn @ w1 + b1)       [T,2048] f16
    eplb_wmma_gemm<128><<<dim3(H / 128, T / 128), 256, 0, stream>>>(
        xn, w1f, b1, hbuf, nullptr, T, H, H, 1.f, 0, 1, 1);
    // lh = relu(xn @ lw1 + lb1)     [T,1024] f16
    eplb_wmma_gemm<128><<<dim3(Hh / 128, T / 128), 256, 0, stream>>>(
        xn, lw1f, lb1, lhbuf, nullptr, T, Hh, H, 1.f, 0, 1, 1);
    // logits  = h @ w2 + b2         [T,64] f32
    eplb_wmma_gemm<64><<<dim3(1, T / 128), 256, 0, stream>>>(
        hbuf, w2f, b2, nullptr, logits, T, E, H, 1.f, 0, 0, 0);
    // logits += 0.01 * (lh @ lw2 + lb2)
    eplb_wmma_gemm<64><<<dim3(1, T / 128), 256, 0, stream>>>(
        lhbuf, lw2f, lb2, nullptr, logits, T, E, Hh, 0.01f, 1, 0, 0);

    // softmax + top-2 + load accumulation
    eplb_zero64<<<1, 64, 0, stream>>>(loadAcc);
    int*   outIdx = (int*)d_out;                       // [T,2] int32
    float* outP   = (float*)d_out + (size_t)2 * T;     // [T,2] f32
    float* auxNl  = (float*)d_out + (size_t)4 * T;     // [1] aux, [64] new_loads
    eplb_softmax_topk<<<T / 8, 256, 0, stream>>>(logits, outIdx, outP, loadAcc);
    eplb_finalize<<<1, 64, 0, stream>>>(eld, loadAcc, auxNl, 1.f / (float)T);
}
